// TransformerEncoderLayer_6811818132412
// MI455X (gfx1250) — matrix operand order repaired and checked by reading, not compiled, not run on hardware
//
#include <hip/hip_runtime.h>
#include <hip/hip_bf16.h>
#include <math.h>

// ---------------------------------------------------------------------------
// Problem constants (reference: B=4, C=64, H=W=128, E=8, groups=2, mlp=64)
// ---------------------------------------------------------------------------
#define BB   4
#define CC   64
#define NN   16384        // H*W
#define EE   8
#define SPLITK 32         // K-split factor for the Gram matrix
#define KCH  (NN / SPLITK)
#define GN_EPS_F 1e-05f
#define L2_EPS_F 1e-12f

typedef __attribute__((ext_vector_type(16))) _Float16 v16h;
typedef __attribute__((ext_vector_type(8)))  _Float16 v8h;
typedef __attribute__((ext_vector_type(8)))  float    v8f;

union V16H { v16h v; _Float16 e[16]; v8h h2[2]; };

// ---------------------------------------------------------------------------
// per-(b,c) channel sums S1=Σx, S2=Σx² over N=16384 pixels.
// ---------------------------------------------------------------------------
__global__ void k_chan_sums(const float* __restrict__ x,
                            float* __restrict__ s1, float* __restrict__ s2) {
  __shared__ float r1[256], r2[256];
  int row = blockIdx.x;                       // b*CC + c
  const float* p = x + (size_t)row * NN;
  float a = 0.f, b = 0.f;
  for (int i = threadIdx.x; i < NN; i += 256) {
    float v = p[i];
    a += v; b += v * v;
  }
  r1[threadIdx.x] = a; r2[threadIdx.x] = b;
  __syncthreads();
  for (int s = 128; s > 0; s >>= 1) {
    if ((int)threadIdx.x < s) {
      r1[threadIdx.x] += r1[threadIdx.x + s];
      r2[threadIdx.x] += r2[threadIdx.x + s];
    }
    __syncthreads();
  }
  if (threadIdx.x == 0) { s1[row] = r1[0]; s2[row] = r2[0]; }
}

// ---------------------------------------------------------------------------
// GroupNorm stats -> per-channel affine (a,s): xn = a*x + s.  Optionally also
// analytic Σxn, Σxn² (for q/k L2 norms without materializing q/k).
// ---------------------------------------------------------------------------
__global__ void k_gn_stats(const float* __restrict__ s1, const float* __restrict__ s2,
                           const float* __restrict__ w,  const float* __restrict__ bia,
                           float* __restrict__ aff, float* __restrict__ sn) {
  __shared__ float gm[BB * 2], gv[BB * 2];
  int tid = threadIdx.x;
  if (tid < BB * 2) {
    int b = tid >> 1, g = tid & 1;
    float S1 = 0.f, S2 = 0.f;
    for (int c = g * 32; c < g * 32 + 32; ++c) {
      S1 += s1[b * CC + c]; S2 += s2[b * CC + c];
    }
    float inv = 1.0f / (32.0f * (float)NN);
    float mu = S1 * inv;
    gm[tid] = mu;
    gv[tid] = S2 * inv - mu * mu;              // population variance
  }
  __syncthreads();
  if (tid < BB * CC) {
    int b = tid >> 6, c = tid & 63, g = c >> 5;
    float mu = gm[b * 2 + g], var = gv[b * 2 + g];
    float a = w[c] * rsqrtf(var + GN_EPS_F);
    float s = bia[c] - mu * a;
    aff[tid * 2 + 0] = a;
    aff[tid * 2 + 1] = s;
    if (sn) {
      float S1 = s1[tid], S2 = s2[tid], Nf = (float)NN;
      sn[tid * 2 + 0] = a * S1 + s * Nf;
      sn[tid * 2 + 1] = a * a * S2 + 2.0f * a * s * S1 + s * s * Nf;
    }
  }
}

// ---------------------------------------------------------------------------
// Materialize xn as packed f16 (half the bytes of f32; L2-resident).
// One block per channel row; float4 in, 4 packed halves (8B) out.
// ---------------------------------------------------------------------------
__global__ void k_xnorm_f16(const float* __restrict__ x, const float* __restrict__ aff,
                            _Float16* __restrict__ xnh) {
  int row = blockIdx.x;
  float a = aff[row * 2 + 0], s = aff[row * 2 + 1];
  const float4* p = (const float4*)(x + (size_t)row * NN);
  unsigned long long* o = (unsigned long long*)(xnh + (size_t)row * NN);
  union { _Float16 h[4]; unsigned long long u; } pk;
  for (int i = threadIdx.x; i < NN / 4; i += 256) {
    float4 v = p[i];
    pk.h[0] = (_Float16)(v.x * a + s);
    pk.h[1] = (_Float16)(v.y * a + s);
    pk.h[2] = (_Float16)(v.z * a + s);
    pk.h[3] = (_Float16)(v.w * a + s);
    o[i] = pk.u;
  }
}

// ---------------------------------------------------------------------------
// Split-K Gram: Gp[chunk][b][c][d] = Σ_{n in chunk} xn[c,n]·xn[d,n] via WMMA.
// 2048 waves (B·16 tiles · 32 K-chunks), 16 WMMA steps each; f16 operands
// loaded as aligned 16B vectors (global_load_b128).
// A layout: lanes0-15 K0-15, lanes16-31 K16-31 (8-half runs).
// B layout: half-waves split K (0-15/16-31), column N = lane%16.
// ---------------------------------------------------------------------------
__global__ void k_gram_split(const _Float16* __restrict__ xnh, float* __restrict__ Gp) {
  int wave = threadIdx.x >> 5, lane = threadIdx.x & 31;
  int w = blockIdx.x * 8 + wave;               // 0..2047
  int chunk = w & (SPLITK - 1);
  int tile  = (w >> 5) & 15;
  int b     = w >> 9;
  int ci = tile >> 2, di = tile & 3;
  int rM = ci * 16 + (lane & 15);
  int rN = di * 16 + (lane & 15);
  const _Float16* xb   = xnh + (size_t)b * CC * NN;
  const _Float16* rowM = xb + (size_t)rM * NN;
  const _Float16* rowN = xb + (size_t)rN * NN;
  int kloA = (lane < 16) ? 0 : 16;
  int kloB = (lane < 16) ? 0 : 16;
  int nbeg = chunk * KCH;
  v8f acc = {};
  for (int n0 = nbeg; n0 < nbeg + KCH; n0 += 32) {
    __builtin_prefetch((const void*)(rowM + n0 + 256), 0, 1);
    V16H a, bo;
    a.h2[0]  = *(const v8h*)(rowM + n0 + kloA);
    a.h2[1]  = *(const v8h*)(rowM + n0 + kloA + 8);
    bo.h2[0] = *(const v8h*)(rowN + n0 + kloB);
    bo.h2[1] = *(const v8h*)(rowN + n0 + kloB + 8);
    acc = __builtin_amdgcn_wmma_f32_16x16x32_f16(false, a.v, false, bo.v,
                                                 (short)0, acc, false, false);
  }
  // D layout: VGPR r -> M = r + 8*(lane>=16), N = lane%16
  int mbase = ci * 16 + ((lane >= 16) ? 8 : 0);
  int ncol  = di * 16 + (lane & 15);
  float* Gb = Gp + ((size_t)chunk * BB + b) * CC * CC;
#pragma unroll
  for (int r = 0; r < 8; ++r)
    Gb[(mbase + r) * CC + ncol] = acc[r];
}

// Deterministic reduction of the SPLITK partial Grams.
__global__ void k_gram_reduce(const float* __restrict__ Gp, float* __restrict__ G) {
  int idx = blockIdx.x * 256 + threadIdx.x;    // [0, BB*CC*CC)
  float s = 0.f;
  for (int k = 0; k < SPLITK; ++k)
    s += Gp[(size_t)k * (BB * CC * CC) + idx];
  G[idx] = s;
}

// ---------------------------------------------------------------------------
// Logits from Gram + analytic L2 norms, softmax -> P[b,e,c,d].
// ---------------------------------------------------------------------------
__global__ void k_attn(const float* __restrict__ G,  const float* __restrict__ qkvw,
                       const float* __restrict__ qkvb, const float* __restrict__ t,
                       const float* __restrict__ sn, float* __restrict__ P) {
  int b = blockIdx.x >> 3, h = blockIdx.x & 7;
  int c = threadIdx.x;
  __shared__ float wk[CC], bk[CC], kn[CC], s1d[CC];
  float Nf = (float)NN;
  float wq  = qkvw[c * 24 + h],      bq  = qkvb[c * 24 + h];
  float wkc = qkvw[c * 24 + 8 + h],  bkc = qkvb[c * 24 + 8 + h];
  float S1 = sn[(b * CC + c) * 2 + 0], S2 = sn[(b * CC + c) * 2 + 1];
  float knc = sqrtf(fmaxf(wkc * wkc * S2 + 2.f * wkc * bkc * S1 + bkc * bkc * Nf, 0.f));
  wk[c] = wkc; bk[c] = bkc; kn[c] = fmaxf(knc, L2_EPS_F); s1d[c] = S1;
  float qn = sqrtf(fmaxf(wq * wq * S2 + 2.f * wq * bq * S1 + bq * bq * Nf, 0.f));
  qn = fmaxf(qn, L2_EPS_F);
  __syncthreads();
  const float* Gb = G + b * CC * CC;
  float th = t[h];
  float l[CC];
  float mx = -1e30f;
  for (int d = 0; d < CC; ++d) {
    float num = wq * wk[d] * Gb[c * CC + d] + wq * bk[d] * S1
              + bq * wk[d] * s1d[d] + bq * bk[d] * Nf;
    float v = th * num / (qn * kn[d]);
    l[d] = v; mx = fmaxf(mx, v);
  }
  float sum = 0.f;
  for (int d = 0; d < CC; ++d) { l[d] = expf(l[d] - mx); sum += l[d]; }
  float inv = 1.f / sum;
  float* Pp = P + ((size_t)(b * EE + h) * CC + c) * CC;
  for (int d = 0; d < CC; ++d) Pp[d] = l[d] * inv;
}

// ---------------------------------------------------------------------------
// Collapse heads -> per-batch 64x64 mix matrix and per-channel constant.
// ---------------------------------------------------------------------------
__global__ void k_mix_accum(const float* __restrict__ P,  const float* __restrict__ qkvw,
                            const float* __restrict__ qkvb, const float* __restrict__ fw,
                            const float* __restrict__ fb, float* __restrict__ Mm,
                            float* __restrict__ cst) {
  int b = blockIdx.x;
  for (int p = threadIdx.x; p < CC * CC; p += 256) {
    int c = p >> 6, d = p & 63;
    float m = 0.f;
#pragma unroll
    for (int e = 0; e < EE; ++e)
      m += fw[c * EE + e] * P[(((size_t)b * EE + e) * CC + c) * CC + d]
                          * qkvw[d * 24 + 16 + e];
    Mm[b * CC * CC + p] = m;
  }
  if (threadIdx.x < CC) {
    int c = threadIdx.x;
    float s = fb[c];
    for (int e = 0; e < EE; ++e) {
      const float* Pp = P + (((size_t)b * EE + e) * CC + c) * CC;
      float acc = 0.f;
      for (int d = 0; d < CC; ++d) acc += Pp[d] * qkvb[d * 24 + 16 + e];
      s += fw[c * EE + e] * acc;
    }
    cst[b * CC + c] = s;
  }
}

// ---------------------------------------------------------------------------
// x2[c,n] = x[c,n] + cst[c] + Σ_d M[c,d]·xn[d,n]  via WMMA.
// 8 waves = 4 c-tiles × 2 n-tiles per block; M staged in LDS as f16; xn read
// directly from the packed f16 field (coalesced across lanes per K element).
// ---------------------------------------------------------------------------
__global__ void k_apply_mix(const float* __restrict__ x, const _Float16* __restrict__ xnh,
                            const float* __restrict__ Mm, const float* __restrict__ cst,
                            float* __restrict__ y) {
  __shared__ _Float16 Msh[CC][CC];
  __shared__ float cstS[CC];
  int b  = blockIdx.x >> 9;                    // NN/32 = 512 blocks per batch
  int n0 = (blockIdx.x & 511) * 32;
  for (int p = threadIdx.x; p < CC * CC; p += 256)
    Msh[p >> 6][p & 63] = (_Float16)Mm[b * CC * CC + p];
  if (threadIdx.x < CC) cstS[threadIdx.x] = cst[b * CC + threadIdx.x];
  __syncthreads();
  int wave = threadIdx.x >> 5, lane = threadIdx.x & 31;
  int ct = wave >> 1, nt = wave & 1;
  int rowc = ct * 16 + (lane & 15);
  int ncol = n0 + nt * 16 + (lane & 15);
  const float*    xb  = x   + (size_t)b * CC * NN;
  const _Float16* xnb = xnh + (size_t)b * CC * NN;
  int kloA = (lane < 16) ? 0 : 16, kloB = (lane < 16) ? 0 : 16;
  v8f acc = {};
#pragma unroll
  for (int k0 = 0; k0 < CC; k0 += 32) {
    V16H a, bo;
#pragma unroll
    for (int j = 0; j < 8; ++j) {
      a.e[j]     = Msh[rowc][k0 + kloA + j];
      a.e[8 + j] = Msh[rowc][k0 + 8 + kloA + j];
    }
#pragma unroll
    for (int j = 0; j < 16; ++j)
      bo.e[j] = xnb[(size_t)(k0 + kloB + j) * NN + ncol];
    acc = __builtin_amdgcn_wmma_f32_16x16x32_f16(false, a.v, false, bo.v,
                                                 (short)0, acc, false, false);
  }
  int mbase = ct * 16 + ((lane >= 16) ? 8 : 0);
  float* yb = y + (size_t)b * CC * NN;
#pragma unroll
  for (int r = 0; r < 8; ++r) {
    int cr = mbase + r;
    yb[(size_t)cr * NN + ncol] = acc[r] + xb[(size_t)cr * NN + ncol] + cstS[cr];
  }
}

// ---------------------------------------------------------------------------
// Fused FFN, channel-major:  out = x2 + W2ᵀ·gelu(W1ᵀ·xn2 + b1) + b2.
// Two chained WMMA GEMMs; hidden 64×32 tile staged in LDS as f16.
// ---------------------------------------------------------------------------
__global__ void k_ffn(const float* __restrict__ x2, const _Float16* __restrict__ xn2h,
                      const float* __restrict__ w1, const float* __restrict__ b1,
                      const float* __restrict__ w2, const float* __restrict__ b2,
                      float* __restrict__ out) {
  __shared__ _Float16 W1sh[CC][CC];            // A1[m][c]    = ff_w1[c][m]
  __shared__ _Float16 W2sh[CC][CC];            // A2[cout][m] = ff_w2[m][cout]
  __shared__ _Float16 Hsh[CC][32];             // hidden tile (f16)
  __shared__ float b1s[CC], b2s[CC];
  int b  = blockIdx.x >> 9;
  int n0 = (blockIdx.x & 511) * 32;
  for (int p = threadIdx.x; p < CC * CC; p += 256) {
    int r = p >> 6, cidx = p & 63;
    W1sh[r][cidx] = (_Float16)w1[cidx * CC + r];
    W2sh[r][cidx] = (_Float16)w2[cidx * CC + r];
  }
  if (threadIdx.x < CC) {
    b1s[threadIdx.x] = b1[threadIdx.x];
    b2s[threadIdx.x] = b2[threadIdx.x];
  }
  __syncthreads();
  int wave = threadIdx.x >> 5, lane = threadIdx.x & 31;
  int rt = wave >> 1, nt = wave & 1;
  int row  = rt * 16 + (lane & 15);
  int nloc = nt * 16 + (lane & 15);
  int ncol = n0 + nloc;
  const float*    xb  = x2   + (size_t)b * CC * NN;
  const _Float16* xnb = xn2h + (size_t)b * CC * NN;
  int kloA = (lane < 16) ? 0 : 16, kloB = (lane < 16) ? 0 : 16;
  int mbase = rt * 16 + ((lane >= 16) ? 8 : 0);

  // Stage 1: H = gelu(W1ᵀ·xn2 + b1)
  v8f acc = {};
#pragma unroll
  for (int k0 = 0; k0 < CC; k0 += 32) {
    V16H a, bo;
#pragma unroll
    for (int j = 0; j < 8; ++j) {
      a.e[j]     = W1sh[row][k0 + kloA + j];
      a.e[8 + j] = W1sh[row][k0 + 8 + kloA + j];
    }
#pragma unroll
    for (int j = 0; j < 16; ++j)
      bo.e[j] = xnb[(size_t)(k0 + kloB + j) * NN + ncol];
    acc = __builtin_amdgcn_wmma_f32_16x16x32_f16(false, a.v, false, bo.v,
                                                 (short)0, acc, false, false);
  }
#pragma unroll
  for (int r = 0; r < 8; ++r) {
    int mm = mbase + r;
    float v = acc[r] + b1s[mm];
    v = 0.5f * v * (1.0f + erff(v * 0.70710678118654752440f));  // exact GELU
    Hsh[mm][nloc] = (_Float16)v;
  }
  __syncthreads();

  // Stage 2: out = x2 + W2ᵀ·H + b2
  v8f acc2 = {};
#pragma unroll
  for (int k0 = 0; k0 < CC; k0 += 32) {
    V16H a, bo;
#pragma unroll
    for (int j = 0; j < 8; ++j) {
      a.e[j]     = W2sh[row][k0 + kloA + j];
      a.e[8 + j] = W2sh[row][k0 + 8 + kloA + j];
    }
#pragma unroll
    for (int j = 0; j < 16; ++j)
      bo.e[j] = Hsh[k0 + kloB + j][nloc];
    acc2 = __builtin_amdgcn_wmma_f32_16x16x32_f16(false, a.v, false, bo.v,
                                                  (short)0, acc2, false, false);
  }
  float* ob = out + (size_t)b * CC * NN;
#pragma unroll
  for (int r = 0; r < 8; ++r) {
    int cr = mbase + r;
    ob[(size_t)cr * NN + ncol] = acc2[r] + b2s[cr] + xb[(size_t)cr * NN + ncol];
  }
}

// ---------------------------------------------------------------------------
// Launch
// ---------------------------------------------------------------------------
extern "C" void kernel_launch(void* const* d_in, const int* in_sizes, int n_in,
                              void* d_out, int out_size, void* d_ws, size_t ws_size,
                              hipStream_t stream) {
  const float* x     = (const float*)d_in[0];
  const float* gn1w  = (const float*)d_in[1];
  const float* gn1b  = (const float*)d_in[2];
  const float* qkvw  = (const float*)d_in[3];
  const float* qkvb  = (const float*)d_in[4];
  const float* t     = (const float*)d_in[5];
  const float* fusw  = (const float*)d_in[6];
  const float* fusb  = (const float*)d_in[7];
  const float* gn2w  = (const float*)d_in[8];
  const float* gn2b  = (const float*)d_in[9];
  const float* ffw1  = (const float*)d_in[10];
  const float* ffb1  = (const float*)d_in[11];
  const float* ffw2  = (const float*)d_in[12];
  const float* ffb2  = (const float*)d_in[13];
  float* out = (float*)d_out;

  // workspace layout (floats first, then packed f16 regions)
  float* w    = (float*)d_ws;
  float* s1   = w;                               // 256
  float* s2   = s1 + BB * CC;                    // 256
  float* aff1 = s2 + BB * CC;                    // 512
  float* sn   = aff1 + BB * CC * 2;              // 512
  float* G    = sn + BB * CC * 2;                // 16384
  float* P    = G + BB * CC * CC;                // 131072
  float* Mm   = P + BB * EE * CC * CC;           // 16384
  float* cst  = Mm + BB * CC * CC;               // 256
  float* x2   = cst + BB * CC;                   // 4,194,304 (16 MB)
  float* s1b  = x2 + (size_t)BB * CC * NN;       // 256
  float* s2b  = s1b + BB * CC;                   // 256
  float* aff2 = s2b + BB * CC;                   // 512
  float* Gp   = aff2 + BB * CC * 2;              // SPLITK*BB*CC*CC = 524288 (2 MB)
  _Float16* xnh  = (_Float16*)(Gp + (size_t)SPLITK * BB * CC * CC);  // 8 MB
  _Float16* xn2h = xnh + (size_t)BB * CC * NN;                        // 8 MB

  k_chan_sums <<<BB * CC, 256, 0, stream>>>(x, s1, s2);
  k_gn_stats  <<<1, 256, 0, stream>>>(s1, s2, gn1w, gn1b, aff1, sn);
  k_xnorm_f16 <<<BB * CC, 256, 0, stream>>>(x, aff1, xnh);
  k_gram_split<<<BB * 16 * SPLITK / 8, 256, 0, stream>>>(xnh, Gp);
  k_gram_reduce<<<BB * CC * CC / 256, 256, 0, stream>>>(Gp, G);
  k_attn      <<<BB * EE, CC, 0, stream>>>(G, qkvw, qkvb, t, sn, P);
  k_mix_accum <<<BB, 256, 0, stream>>>(P, qkvw, qkvb, fusw, fusb, Mm, cst);
  k_apply_mix <<<BB * (NN / 32), 256, 0, stream>>>(x, xnh, Mm, cst, x2);
  k_chan_sums <<<BB * CC, 256, 0, stream>>>(x2, s1b, s2b);
  k_gn_stats  <<<1, 256, 0, stream>>>(s1b, s2b, gn2w, gn2b, aff2, nullptr);
  k_xnorm_f16 <<<BB * CC, 256, 0, stream>>>(x2, aff2, xn2h);
  k_ffn       <<<BB * (NN / 32), 256, 0, stream>>>(x2, xn2h, ffw1, ffb1, ffw2, ffb2, out);
}
